// BasicRNNLayer_20933670601159
// MI455X (gfx1250) — matrix operand order repaired and checked by reading, not compiled, not run on hardware
//
#include <hip/hip_runtime.h>

// Vanilla RNN  h_t = relu([x_t, h_{t-1}] @ W + b), B=64 T=512 D=H=1024.
// MI455X strategy:
//   (0) split_w: one-time transpose+split of Wx into bf16 (hi, lo) stored in
//       WMMA B-fragment order (column-major) in d_ws (if ws_size allows).
//   (1) pre = X @ Wx + b : [32768 x 1024 x 1024] GEMM, bf16x3
//       (hi*hi + hi*lo + lo*hi) on V_WMMA_F32_16X16X32_BF16, f32 accum.
//       Wave tile 32x64 (2x4 fragments), block tile 64x256.
//   (2) persistent recurrence (ONE launch, 512 steps): exact f32 WMMA,
//       weight-stationary Wh in VGPRs, K split 4-ways + LDS reduction,
//       L2-atomic grid barrier between steps.

typedef __attribute__((ext_vector_type(2)))  float  v2f;
typedef __attribute__((ext_vector_type(8)))  float  v8f;
typedef __attribute__((ext_vector_type(16))) __bf16 v16bf;

#define RNN_B 64
#define RNN_T 512
#define RNN_D 1024
#define RNN_H 1024

#define WMMA_F32(a, b, c) \
  __builtin_amdgcn_wmma_f32_16x16x4_f32(false, (a), false, (b), (short)0, (c), false, false)
#define WMMA_BF16(a, b, c) \
  __builtin_amdgcn_wmma_f32_16x16x32_bf16(false, (a), false, (b), (short)0, (c), false, false)

// ---------------------------------------------------------------------------
// Kernel 0: Wt_hi[n][k] = bf16(W[k][n]); Wt_lo[n][k] = bf16(W[k][n] - hi).
// LDS-tiled transpose, grid (64, 64) x 256 threads. One-time cost (~4 MB).
// Column-major bf16 == exactly the 16 contiguous K values (this code's k = 16*half + s order) a B-fragment
// lane needs, so the GEMM B-operand becomes two b128 loads per fragment.
// ---------------------------------------------------------------------------
__global__ __launch_bounds__(256) void split_w_kernel(
    const float* __restrict__ W,      // [D+H, H]; rows 0..D-1 = Wx
    __bf16* __restrict__ Whi,         // [H][D] col-major of Wx
    __bf16* __restrict__ Wlo)
{
  __shared__ float tile[16][17];
  const int tx = threadIdx.x & 15, ty = threadIdx.x >> 4;
  const int k0 = blockIdx.x * 16, n0 = blockIdx.y * 16;

  tile[ty][tx] = W[(size_t)(k0 + ty) * RNN_H + n0 + tx];
  __syncthreads();

  const float  f = tile[tx][ty];                 // W[k0+tx][n0+ty]
  const __bf16 h = (__bf16)f;
  const size_t o = (size_t)(n0 + ty) * RNN_D + k0 + tx;
  Whi[o] = h;
  Wlo[o] = (__bf16)(f - (float)h);
}

// ---------------------------------------------------------------------------
// Kernel 1 (fast path): pre = X @ Wx + b with pre-split B.
// 8 wave32s as 2(M) x 4(N); wave tile 32x64; block tile 64x256; grid 512x4.
// A fragment (16x32 bf16): lane = M row; 16 contiguous K at k0+half*16,
// split inline. B fragment: v16bf straight from Wt_{hi,lo}.
// ---------------------------------------------------------------------------
__global__ __launch_bounds__(256) void rnn_precompute_fast_kernel(
    const float*  __restrict__ X,     // [B*T, D]
    const __bf16* __restrict__ Whi,   // [H][D] col-major
    const __bf16* __restrict__ Wlo,
    const float*  __restrict__ bias,  // [H]
    float* __restrict__ Out)          // [B*T, H]
{
  const int lane = threadIdx.x & 31;
  const int wid  = threadIdx.x >> 5;
  const int half = lane >> 4;
  const int lid  = lane & 15;

  const int m0 = blockIdx.x * 64  + (wid >> 2) * 32;
  const int n0 = blockIdx.y * 256 + (wid & 3) * 64;

  v8f acc[2][4] = {};

  const float* Arow0 = X + (size_t)(m0 + lid) * RNN_D;
  const float* Arow1 = Arow0 + (size_t)16 * RNN_D;
  // Per-lane B bases: column n0 + nt*16 + lid, contiguous in K.
  const __bf16* Bh[4];
  const __bf16* Bl[4];
#pragma unroll
  for (int nt = 0; nt < 4; ++nt) {
    const size_t col = (size_t)(n0 + nt * 16 + lid) * RNN_D;
    Bh[nt] = Whi + col;
    Bl[nt] = Wlo + col;
  }

  for (int k = 0; k < RNN_D; k += 32) {
    // ---- A fragments: load f32, split to (hi, lo) bf16
    v16bf a0h, a0l, a1h, a1l;
    {
      const float* p00 = Arow0 + k + half * 16;
      const float* p01 = p00 + 8;
      const float* p10 = Arow1 + k + half * 16;
      const float* p11 = p10 + 8;
#pragma unroll
      for (int i = 0; i < 8; ++i) {
        float f0 = p00[i], f1 = p01[i], f2 = p10[i], f3 = p11[i];
        __bf16 h0 = (__bf16)f0, h1 = (__bf16)f1, h2 = (__bf16)f2, h3 = (__bf16)f3;
        a0h[i] = h0; a0h[i + 8] = h1;
        a1h[i] = h2; a1h[i + 8] = h3;
        a0l[i]     = (__bf16)(f0 - (float)h0);
        a0l[i + 8] = (__bf16)(f1 - (float)h1);
        a1l[i]     = (__bf16)(f2 - (float)h2);
        a1l[i + 8] = (__bf16)(f3 - (float)h3);
      }
    }
    // ---- B fragments: direct vector loads (no conversion) + bf16x3 WMMA
    const int kb = k + half * 16;
#pragma unroll
    for (int nt = 0; nt < 4; ++nt) {
      v16bf bh = *(const v16bf*)(Bh[nt] + kb);
      v16bf bl = *(const v16bf*)(Bl[nt] + kb);
      acc[0][nt] = WMMA_BF16(a0h, bh, acc[0][nt]);
      acc[0][nt] = WMMA_BF16(a0h, bl, acc[0][nt]);
      acc[0][nt] = WMMA_BF16(a0l, bh, acc[0][nt]);
      acc[1][nt] = WMMA_BF16(a1h, bh, acc[1][nt]);
      acc[1][nt] = WMMA_BF16(a1h, bl, acc[1][nt]);
      acc[1][nt] = WMMA_BF16(a1l, bh, acc[1][nt]);
    }
  }

  float bb[4];
#pragma unroll
  for (int nt = 0; nt < 4; ++nt) bb[nt] = bias[n0 + nt * 16 + lid];
#pragma unroll
  for (int mt = 0; mt < 2; ++mt) {
    float* Orow = Out + (size_t)(m0 + mt * 16 + half * 8) * RNN_H + n0 + lid;
#pragma unroll
    for (int r = 0; r < 8; ++r) {
#pragma unroll
      for (int nt = 0; nt < 4; ++nt)
        Orow[(size_t)r * RNN_H + nt * 16] = acc[mt][nt][r] + bb[nt];
    }
  }
}

// ---------------------------------------------------------------------------
// Kernel 1 (fallback, no workspace): inline-split bf16x3, wave tile 32x32.
// ---------------------------------------------------------------------------
__global__ __launch_bounds__(256) void rnn_precompute_kernel(
    const float* __restrict__ X,
    const float* __restrict__ W,
    const float* __restrict__ bias,
    float* __restrict__ Out)
{
  const int lane = threadIdx.x & 31;
  const int wid  = threadIdx.x >> 5;
  const int half = lane >> 4;
  const int lid  = lane & 15;

  const int m0 = blockIdx.x * 64  + (wid >> 2) * 32;
  const int n0 = blockIdx.y * 128 + (wid & 3) * 32;

  v8f c00 = {}, c01 = {}, c10 = {}, c11 = {};

  const float* Arow0 = X + (size_t)(m0 + lid) * RNN_D;
  const float* Arow1 = Arow0 + (size_t)16 * RNN_D;
  const float* Bcol0 = W + n0 + lid;
  const float* Bcol1 = Bcol0 + 16;

  for (int k = 0; k < RNN_D; k += 32) {
    v16bf a0h, a0l, a1h, a1l;
    {
      const float* p00 = Arow0 + k + half * 16;
      const float* p01 = p00 + 8;
      const float* p10 = Arow1 + k + half * 16;
      const float* p11 = p10 + 8;
#pragma unroll
      for (int i = 0; i < 8; ++i) {
        float f0 = p00[i], f1 = p01[i], f2 = p10[i], f3 = p11[i];
        __bf16 h0 = (__bf16)f0, h1 = (__bf16)f1, h2 = (__bf16)f2, h3 = (__bf16)f3;
        a0h[i] = h0; a0h[i + 8] = h1;
        a1h[i] = h2; a1h[i + 8] = h3;
        a0l[i]     = (__bf16)(f0 - (float)h0);
        a0l[i + 8] = (__bf16)(f1 - (float)h1);
        a1l[i]     = (__bf16)(f2 - (float)h2);
        a1l[i + 8] = (__bf16)(f3 - (float)h3);
      }
    }
    v16bf b0h, b0l, b1h, b1l;
    {
      const float* q0 = Bcol0 + (size_t)(k + half * 16) * RNN_H;
      const float* q1 = Bcol1 + (size_t)(k + half * 16) * RNN_H;
#pragma unroll
      for (int j = 0; j < 16; ++j) {
        float f0 = q0[(size_t)j * RNN_H];
        float f1 = q1[(size_t)j * RNN_H];
        __bf16 h0 = (__bf16)f0, h1 = (__bf16)f1;
        b0h[j] = h0; b1h[j] = h1;
        b0l[j] = (__bf16)(f0 - (float)h0);
        b1l[j] = (__bf16)(f1 - (float)h1);
      }
    }
    c00 = WMMA_BF16(a0h, b0h, c00);
    c00 = WMMA_BF16(a0h, b0l, c00);
    c00 = WMMA_BF16(a0l, b0h, c00);
    c01 = WMMA_BF16(a0h, b1h, c01);
    c01 = WMMA_BF16(a0h, b1l, c01);
    c01 = WMMA_BF16(a0l, b1h, c01);
    c10 = WMMA_BF16(a1h, b0h, c10);
    c10 = WMMA_BF16(a1h, b0l, c10);
    c10 = WMMA_BF16(a1l, b0h, c10);
    c11 = WMMA_BF16(a1h, b1h, c11);
    c11 = WMMA_BF16(a1h, b1l, c11);
    c11 = WMMA_BF16(a1l, b1h, c11);
  }

  const float bb0 = bias[n0 + lid];
  const float bb1 = bias[n0 + 16 + lid];
  v8f acc[2][2] = {{c00, c01}, {c10, c11}};
#pragma unroll
  for (int mt = 0; mt < 2; ++mt) {
    float* Orow = Out + (size_t)(m0 + mt * 16 + half * 8) * RNN_H + n0 + lid;
#pragma unroll
    for (int r = 0; r < 8; ++r) {
      Orow[(size_t)r * RNN_H]      = acc[mt][0][r] + bb0;
      Orow[(size_t)r * RNN_H + 16] = acc[mt][1][r] + bb1;
    }
  }
}

// ---------------------------------------------------------------------------
// Kernel 2: persistent recurrence (native f32 WMMA, exact).
// Grid: (32 N-slices) x (4 M-slices) = 128 blocks, 256 threads each.
// ---------------------------------------------------------------------------
#define KCHUNK 256
#define NBLOCKS 128u

__global__ __launch_bounds__(256) void rnn_recurrence_kernel(
    float* __restrict__ Out,
    const float* __restrict__ Wh,
    unsigned* __restrict__ bar)
{
  const int lane = threadIdx.x & 31;
  const int wid  = threadIdx.x >> 5;
  const int half = lane >> 4;
  const int lid  = lane & 15;

  const int nw    = wid & 1;
  const int kw    = wid >> 1;
  const int n0    = blockIdx.x * 32 + nw * 16;
  const int mbase = blockIdx.y * 16;
  const int kbase = kw * KCHUNK;

  v2f breg[KCHUNK / 4];
  {
    const float* Bcol = Wh + n0 + lid;
#pragma unroll
    for (int kk = 0; kk < KCHUNK / 4; ++kk) {
      const int k = kbase + kk * 4 + half * 2;
      breg[kk].x = Bcol[(size_t)k * RNN_H];
      breg[kk].y = Bcol[(size_t)(k + 1) * RNN_H];
    }
  }

  __shared__ float red[8][16][16];
  const size_t bstride = (size_t)RNN_T * RNN_H;

  for (int t = 0; t < RNN_T; ++t) {
    v8f c = {};
    if (t > 0) {
      const float* Arow =
          Out + (size_t)(mbase + lid) * bstride + (size_t)(t - 1) * RNN_H + kbase;
#pragma unroll
      for (int kk = 0; kk < KCHUNK / 4; ++kk) {
        v2f a = *(const v2f*)(Arow + kk * 4 + half * 2);
        c = WMMA_F32(a, breg[kk], c);
      }
    }

#pragma unroll
    for (int r = 0; r < 8; ++r) red[wid][half * 8 + r][lid] = c[r];
    __syncthreads();

    if (kw == 0) {
      float* Orow =
          Out + (size_t)(mbase + half * 8) * bstride + (size_t)t * RNN_H + n0 + lid;
#pragma unroll
      for (int r = 0; r < 8; ++r) {
        const int row = half * 8 + r;
        float v = Orow[(size_t)r * bstride]
                + red[nw][row][lid]     + red[nw + 2][row][lid]
                + red[nw + 4][row][lid] + red[nw + 6][row][lid];
        Orow[(size_t)r * bstride] = v > 0.0f ? v : 0.0f;
      }
      __threadfence();
    }
    __syncthreads();

    if (t < RNN_T - 1) {
      if (threadIdx.x == 0) {
        __hip_atomic_fetch_add(bar, 1u, __ATOMIC_ACQ_REL, __HIP_MEMORY_SCOPE_AGENT);
        const unsigned target = (unsigned)(t + 1) * NBLOCKS;
        while (__hip_atomic_load(bar, __ATOMIC_ACQUIRE, __HIP_MEMORY_SCOPE_AGENT)
               < target) {
          __builtin_amdgcn_s_sleep(1);
        }
      }
      __syncthreads();
    }
  }
}

// ---------------------------------------------------------------------------
extern "C" void kernel_launch(void* const* d_in, const int* in_sizes, int n_in,
                              void* d_out, int out_size, void* d_ws, size_t ws_size,
                              hipStream_t stream) {
  (void)in_sizes; (void)n_in; (void)out_size;

  const float* x    = (const float*)d_in[0];  // [B, T, D] f32
  const float* W    = (const float*)d_in[1];  // [D+H, H] f32
  const float* bias = (const float*)d_in[2];  // [H] f32
  float*       out  = (float*)d_out;          // [B, T, H] f32

  // Workspace layout: [0,4) barrier counter; [256, 256+2MB) Wt_hi; then Wt_lo.
  unsigned* bar = (unsigned*)d_ws;
  const size_t whalf = (size_t)RNN_D * RNN_H * sizeof(__bf16);   // 2 MB
  const size_t need  = 256 + 2 * whalf;
  __bf16* Whi = (__bf16*)((char*)d_ws + 256);
  __bf16* Wlo = (__bf16*)((char*)d_ws + 256 + whalf);

  hipMemsetAsync(bar, 0, sizeof(unsigned), stream);

  if (ws_size >= need) {
    split_w_kernel<<<dim3(RNN_D / 16, RNN_H / 16), 256, 0, stream>>>(W, Whi, Wlo);
    rnn_precompute_fast_kernel<<<dim3((RNN_B * RNN_T) / 64, RNN_H / 256), 256, 0,
                                 stream>>>(x, Whi, Wlo, bias, out);
  } else {
    rnn_precompute_kernel<<<dim3((RNN_B * RNN_T) / 64, RNN_H / 128), 256, 0,
                            stream>>>(x, W, bias, out);
  }

  rnn_recurrence_kernel<<<dim3(RNN_H / 32, RNN_B / 16), 256, 0, stream>>>(
      out, W + (size_t)RNN_D * RNN_H, bar);
}
